// MORTM_48206712930971
// MI455X (gfx1250) — matrix operand order repaired and checked by reading, not compiled, not run on hardware
//
#include <hip/hip_runtime.h>

// ---------------------------------------------------------------------------
// MI455X (gfx1250) Music-Transformer forward.
// bf16 WMMA (v_wmma_f32_16x16x32_bf16) for all matmuls; flash-style attention
// so the SxS score tensor never touches HBM; residual stream kept in f32.
// ---------------------------------------------------------------------------

typedef unsigned short bf16_t;
typedef __bf16 bf16x16 __attribute__((ext_vector_type(16)));
typedef float  f32x8   __attribute__((ext_vector_type(8)));
typedef unsigned int u32x4 __attribute__((ext_vector_type(4)));

#define DEVI __device__ __forceinline__

constexpr int S_  = 2048;
constexpr int B_  = 2;
constexpr int D_  = 512;
constexpr int H_  = 8;
constexpr int HD_ = 64;
constexpr int FF_ = 1024;
constexpr int L_  = 6;
constexpr int V_  = 267;
constexpr float NEG_ = -1e9f;

DEVI bf16_t f2bf(float f) {
  unsigned u = __float_as_uint(f);
  unsigned r = u + 0x7FFFu + ((u >> 16) & 1u);   // round-to-nearest-even
  return (bf16_t)(r >> 16);
}
DEVI float bf2f(bf16_t h) { return __uint_as_float(((unsigned)h) << 16); }

union frag_u { bf16x16 v; u32x4 q[2]; };

DEVI f32x8 zero8() { f32x8 z = {0.f,0.f,0.f,0.f,0.f,0.f,0.f,0.f}; return z; }

// A fragment, 16x32 bf16, row-major source with leading dim ld.
// lane<16: row=lane,  K: 0..15 (contiguous)
// lane>=16: same rows, K: 16..31 (matches B fragment order)
DEVI bf16x16 load_A16x32(const bf16_t* base, int ld, int lane) {
  const int row = lane & 15, kb = (lane >> 4) * 16;
  frag_u u;
  const bf16_t* p = base + (size_t)row * ld + kb;
  u.q[0] = *(const u32x4*)(p);
  u.q[1] = *(const u32x4*)(p + 8);
  return u.v;
}

// B fragment, 32x16: B[k][n] = src[n][k] (src row-major, leading dim ld).
// lane&15 selects column n (a row of src); lane half selects K 0..15 / 16..31.
DEVI bf16x16 load_B32x16(const bf16_t* base, int ld, int lane) {
  const int col = lane & 15, kb = (lane >> 4) * 16;
  frag_u u;
  const bf16_t* p = base + (size_t)col * ld + kb;
  u.q[0] = *(const u32x4*)(p);
  u.q[1] = *(const u32x4*)(p + 8);
  return u.v;
}

// B fragment from weight matrix W[N,K] (row-major), clamping the row index.
DEVI bf16x16 load_B_rowclamp(const bf16_t* W, int row0, int N, int ld, int k0, int lane) {
  int row = row0 + (lane & 15);
  row = (row < N) ? row : (N - 1);
  const int kb = (lane >> 4) * 16;
  frag_u u;
  const bf16_t* p = W + (size_t)row * ld + k0 + kb;
  u.q[0] = *(const u32x4*)(p);
  u.q[1] = *(const u32x4*)(p + 8);
  return u.v;
}

DEVI f32x8 wmma_bf16(bf16x16 a, bf16x16 b, f32x8 c) {
  return __builtin_amdgcn_wmma_f32_16x16x32_bf16(
      /*neg_a=*/false, a, /*neg_b=*/false, b,
      /*c_mod=*/(short)0, c, /*reuse_a=*/false, /*reuse_b=*/false);
}

// ---------------------------------------------------------------------------
// Weight conversion f32 -> bf16
// ---------------------------------------------------------------------------
__global__ __launch_bounds__(256) void cvt_bf16_kernel(const float* __restrict__ in,
                                                       bf16_t* __restrict__ out, int n) {
  int t = blockIdx.x * 256 + threadIdx.x;
  if (t < n) out[t] = f2bf(in[t]);
}

// ---------------------------------------------------------------------------
// Embedding gather + sinusoidal positional encoding -> x f32 + bf16 mirror
// ---------------------------------------------------------------------------
__global__ __launch_bounds__(256) void embed_pe_kernel(const int* __restrict__ ids,
                                                       const float* __restrict__ emb,
                                                       float* __restrict__ x,
                                                       bf16_t* __restrict__ xb) {
  int t = blockIdx.x * 256 + threadIdx.x;         // [0, S*B*D)
  int d = t & (D_ - 1);
  int r = t >> 9;                                  // row = s*B + b
  int b = r & (B_ - 1);
  int s = r >> 1;
  int tok = ids[b * S_ + s];
  int m = d >> 1;
  float ang = (float)s * __expf((float)(2 * m) * (-9.2103403719761836f / (float)D_));
  float pe = (d & 1) ? __cosf(ang) : __sinf(ang);
  float v = emb[(size_t)tok * D_ + d] + pe;
  x[t] = v;
  xb[t] = f2bf(v);
}

// ---------------------------------------------------------------------------
// Generic bf16 GEMM: C[M,N] = A[M,K] * W[N,K]^T + bias, optional ReLU,
// optional f32 and/or bf16 outputs. 256 threads = 8 waves, 128x64 block tile,
// 32x32 per wave (2x2 WMMA tiles), K stepped by 32.
// ---------------------------------------------------------------------------
__global__ __launch_bounds__(256) void gemm_bf16_kernel(
    const bf16_t* __restrict__ A, const bf16_t* __restrict__ W,
    const float* __restrict__ bias, float* __restrict__ outF,
    bf16_t* __restrict__ outB, int M, int N, int K, int relu) {
  const int lane = threadIdx.x & 31;
  const int wv   = threadIdx.x >> 5;
  const int m_base = blockIdx.y * 128 + (wv & 3) * 32;
  const int n_base = blockIdx.x * 64 + (wv >> 2) * 32;

  f32x8 acc[2][2] = {{zero8(), zero8()}, {zero8(), zero8()}};
  const bf16_t* a_row0 = A + (size_t)m_base * K;
  const bf16_t* a_row1 = A + (size_t)(m_base + 16) * K;
  const bf16_t* w_row  = W + (size_t)n_base * K;

  for (int k0 = 0; k0 < K; k0 += 32) {
    bf16x16 a0 = load_A16x32(a_row0 + k0, K, lane);
    bf16x16 a1 = load_A16x32(a_row1 + k0, K, lane);
    bf16x16 b0 = load_B_rowclamp(W, n_base,      N, K, k0, lane);
    bf16x16 b1 = load_B_rowclamp(W, n_base + 16, N, K, k0, lane);
    // WGP-scope prefetch of the next k-chunk (locality 3 -> near caches);
    // speculative, so harmlessly dropped past the end of the buffers.
    __builtin_prefetch((const void*)(a_row0 + k0 + 32), 0, 3);
    __builtin_prefetch((const void*)(w_row  + k0 + 32), 0, 3);
    acc[0][0] = wmma_bf16(a0, b0, acc[0][0]);
    acc[0][1] = wmma_bf16(a0, b1, acc[0][1]);
    acc[1][0] = wmma_bf16(a1, b0, acc[1][0]);
    acc[1][1] = wmma_bf16(a1, b1, acc[1][1]);
  }

  const int col = lane & 15, half = lane >> 4;
#pragma unroll
  for (int ti = 0; ti < 2; ++ti)
#pragma unroll
    for (int tj = 0; tj < 2; ++tj)
#pragma unroll
      for (int v = 0; v < 8; ++v) {
        int gr = m_base + ti * 16 + v + half * 8;
        int gc = n_base + tj * 16 + col;
        if (gc < N) {
          float val = acc[ti][tj][v] + (bias ? bias[gc] : 0.f);
          if (relu) val = fmaxf(val, 0.f);
          if (outF) outF[(size_t)gr * N + gc] = val;
          if (outB) outB[(size_t)gr * N + gc] = f2bf(val);
        }
      }
}

// ---------------------------------------------------------------------------
// Split QKV [SB,3D] -> q (scaled, [B,H,S,HD]), k ([B,H,S,HD]), v^T ([B,H,HD,S])
// ---------------------------------------------------------------------------
__global__ __launch_bounds__(256) void split_qkv_kernel(const bf16_t* __restrict__ qkv,
                                                        bf16_t* __restrict__ qb,
                                                        bf16_t* __restrict__ kb,
                                                        bf16_t* __restrict__ vt) {
  int t = blockIdx.x * 256 + threadIdx.x;          // [0, B*H*S*HD)
  int d = t & (HD_ - 1);
  int s = (t >> 6) & (S_ - 1);
  int h = (t >> 17) & (H_ - 1);
  int b = t >> 20;
  int r = s * B_ + b;
  const bf16_t* row = qkv + (size_t)r * (3 * D_);
  int bh = b * H_ + h;
  float qv = bf2f(row[h * HD_ + d]) * 0.125f;      // 1/sqrt(HD)
  qb[((size_t)bh * S_ + s) * HD_ + d] = f2bf(qv);
  kb[((size_t)bh * S_ + s) * HD_ + d] = row[D_ + h * HD_ + d];
  vt[((size_t)bh * HD_ + d) * S_ + s] = row[2 * D_ + h * HD_ + d];
}

// ---------------------------------------------------------------------------
// Flash attention with Music-Transformer relative positions.
// One wave (32 threads) per (bh, 16-row query tile). Keys in blocks of 32.
// scores = Q K^T (WMMA) + gather of QE = Q Er^T (WMMA) along the skew
// diagonal, causal mask, online softmax, O += P V (WMMA, P via LDS relayout).
// ---------------------------------------------------------------------------
__global__ __launch_bounds__(32) void flash_attn_kernel(
    const bf16_t* __restrict__ qb, const bf16_t* __restrict__ kbuf,
    const bf16_t* __restrict__ vt, const bf16_t* __restrict__ er,
    bf16_t* __restrict__ ctxb) {
  const int lane = threadIdx.x & 31;
  const int qt = blockIdx.x;                    // query tile 0..127
  const int bh = blockIdx.y;                    // 0..15
  const int b = bh >> 3, h = bh & 7;
  const int i0 = qt * 16;

  const bf16_t* Q  = qb   + ((size_t)bh * S_ + i0) * HD_;
  const bf16_t* Kp = kbuf + (size_t)bh * S_ * HD_;
  const bf16_t* Vp = vt   + (size_t)bh * HD_ * S_;

  __shared__ __align__(16) float  qe_lds[16 * 48];
  __shared__ __align__(16) bf16_t p_lds[16 * 32];

  bf16x16 qa0 = load_A16x32(Q, HD_, lane);       // K dims 0..31
  bf16x16 qa1 = load_A16x32(Q + 32, HD_, lane);  // K dims 32..63

  f32x8 o0 = zero8(), o1 = zero8(), o2 = zero8(), o3 = zero8();
  float rmax[8], rsum[8];
#pragma unroll
  for (int v = 0; v < 8; ++v) { rmax[v] = -3.0e38f; rsum[v] = 0.f; }

  const int col = lane & 15, half = lane >> 4;
  const int nblk = (i0 + 47) >> 5;               // cover keys 0..i0+15

  for (int blk = 0; blk < nblk; ++blk) {
    const int j0 = blk * 32;

    // ---- scores: two 16x16 tiles over 32 keys ----
    bf16x16 k00 = load_B32x16(Kp + (size_t)j0 * HD_,             HD_, lane);
    bf16x16 k01 = load_B32x16(Kp + (size_t)j0 * HD_ + 32,        HD_, lane);
    bf16x16 k10 = load_B32x16(Kp + (size_t)(j0 + 16) * HD_,      HD_, lane);
    bf16x16 k11 = load_B32x16(Kp + (size_t)(j0 + 16) * HD_ + 32, HD_, lane);
    f32x8 c0 = zero8(), c1 = zero8();
    c0 = wmma_bf16(qa0, k00, c0); c0 = wmma_bf16(qa1, k01, c0);
    c1 = wmma_bf16(qa0, k10, c1); c1 = wmma_bf16(qa1, k11, c1);

    // ---- QE tiles: er rows ebase..ebase+47 ; e(i',j'') = base0 + j'' - i' ----
    const int ebase = (S_ - 1) - i0 + j0 - 15;
#pragma unroll
    for (int t = 0; t < 3; ++t) {
      int erow = ebase + t * 16 + col;
      erow = erow < 0 ? 0 : (erow > S_ - 1 ? S_ - 1 : erow);   // clamped rows unused
      frag_u u0, u1;
      const bf16_t* p0 = er + (size_t)erow * HD_ + half * 16;  // K dims 0..31
      u0.q[0] = *(const u32x4*)(p0);
      u0.q[1] = *(const u32x4*)(p0 + 8);
      const bf16_t* p1 = p0 + 32;                              // K dims 32..63
      u1.q[0] = *(const u32x4*)(p1);
      u1.q[1] = *(const u32x4*)(p1 + 8);
      f32x8 qe = zero8();
      qe = wmma_bf16(qa0, u0.v, qe);
      qe = wmma_bf16(qa1, u1.v, qe);
#pragma unroll
      for (int v = 0; v < 8; ++v)
        qe_lds[(v + half * 8) * 48 + t * 16 + col] = qe[v];
    }
    asm volatile("s_wait_dscnt 0" ::: "memory");

    // ---- skew gather + causal mask + online softmax ----
#pragma unroll
    for (int v = 0; v < 8; ++v) {
      const int ip = v + half * 8;
      const int gi = i0 + ip;
      float s0 = c0[v] + qe_lds[ip * 48 + (15 + col - ip)];
      if (j0 + col > gi) s0 = NEG_;
      float s1 = c1[v] + qe_lds[ip * 48 + (31 + col - ip)];
      if (j0 + 16 + col > gi) s1 = NEG_;

      float tmax = fmaxf(s0, s1);
#pragma unroll
      for (int m = 1; m < 16; m <<= 1) tmax = fmaxf(tmax, __shfl_xor(tmax, m, 32));
      float mnew = fmaxf(rmax[v], tmax);
      float p0 = __expf(s0 - mnew);
      float p1 = __expf(s1 - mnew);
      float ps = p0 + p1;
#pragma unroll
      for (int m = 1; m < 16; m <<= 1) ps += __shfl_xor(ps, m, 32);
      float corr = __expf(rmax[v] - mnew);
      rsum[v] = rsum[v] * corr + ps;
      rmax[v] = mnew;
      o0[v] *= corr; o1[v] *= corr; o2[v] *= corr; o3[v] *= corr;
      p_lds[ip * 32 + col]      = f2bf(p0);
      p_lds[ip * 32 + 16 + col] = f2bf(p1);
    }
    asm volatile("s_wait_dscnt 0" ::: "memory");

    // ---- O += P * V ; P relayout via LDS, V^T gives contiguous B-fragments ----
    bf16x16 pa = load_A16x32((const bf16_t*)p_lds, 32, lane);
    bf16x16 vb0 = load_B32x16(Vp + (size_t)(0 * 16) * S_ + j0, S_, lane);
    bf16x16 vb1 = load_B32x16(Vp + (size_t)(1 * 16) * S_ + j0, S_, lane);
    bf16x16 vb2 = load_B32x16(Vp + (size_t)(2 * 16) * S_ + j0, S_, lane);
    bf16x16 vb3 = load_B32x16(Vp + (size_t)(3 * 16) * S_ + j0, S_, lane);
    o0 = wmma_bf16(pa, vb0, o0);
    o1 = wmma_bf16(pa, vb1, o1);
    o2 = wmma_bf16(pa, vb2, o2);
    o3 = wmma_bf16(pa, vb3, o3);
  }

  // ---- finalize: ctx[(s,b), h*64 + d] = O / rowsum ----
#pragma unroll
  for (int v = 0; v < 8; ++v) {
    const int ip = v + half * 8;
    const size_t r = (size_t)(i0 + ip) * B_ + b;
    const float inv = 1.0f / rsum[v];
    bf16_t* dst = ctxb + r * D_ + h * HD_ + col;
    dst[0]  = f2bf(o0[v] * inv);
    dst[16] = f2bf(o1[v] * inv);
    dst[32] = f2bf(o2[v] * inv);
    dst[48] = f2bf(o3[v] * inv);
  }
}

// ---------------------------------------------------------------------------
// x = LN(x + res) * g + b  (res nullable). One 256-thread block per row, D=512.
// Writes f32 stream in place and bf16 mirror.
// ---------------------------------------------------------------------------
__global__ __launch_bounds__(256) void add_ln_kernel(float* __restrict__ x,
                                                     const float* __restrict__ res,
                                                     const float* __restrict__ g,
                                                     const float* __restrict__ bta,
                                                     bf16_t* __restrict__ xb) {
  const int r = blockIdx.x;
  const int tid = threadIdx.x;
  float* row = x + (size_t)r * D_;
  float v0 = row[tid], v1 = row[tid + 256];
  if (res) {
    const float* rr = res + (size_t)r * D_;
    v0 += rr[tid]; v1 += rr[tid + 256];
  }
  float s = v0 + v1, s2 = v0 * v0 + v1 * v1;
#pragma unroll
  for (int m = 1; m < 32; m <<= 1) { s += __shfl_xor(s, m, 32); s2 += __shfl_xor(s2, m, 32); }
  __shared__ float sh1[8], sh2[8];
  if ((tid & 31) == 0) { sh1[tid >> 5] = s; sh2[tid >> 5] = s2; }
  __syncthreads();
  float ts = 0.f, ts2 = 0.f;
#pragma unroll
  for (int i = 0; i < 8; ++i) { ts += sh1[i]; ts2 += sh2[i]; }
  float mean = ts * (1.0f / (float)D_);
  float var  = ts2 * (1.0f / (float)D_) - mean * mean;
  float rstd = rsqrtf(var + 1e-5f);
  float o0 = (v0 - mean) * rstd * g[tid] + bta[tid];
  float o1 = (v1 - mean) * rstd * g[tid + 256] + bta[tid + 256];
  row[tid] = o0; row[tid + 256] = o1;
  bf16_t* rb = xb + (size_t)r * D_;
  rb[tid] = f2bf(o0); rb[tid + 256] = f2bf(o1);
}

// ---------------------------------------------------------------------------
// Host-side orchestration
// ---------------------------------------------------------------------------
extern "C" void kernel_launch(void* const* d_in, const int* in_sizes, int n_in,
                              void* d_out, int out_size, void* d_ws, size_t ws_size,
                              hipStream_t stream) {
  (void)in_sizes; (void)n_in; (void)out_size; (void)ws_size;
  const int*   ids  = (const int*)  d_in[0];
  const float* emb  = (const float*)d_in[1];
  const float* inw  = (const float*)d_in[2];
  const float* inb  = (const float*)d_in[3];
  const float* outw = (const float*)d_in[4];
  const float* outb = (const float*)d_in[5];
  const float* er   = (const float*)d_in[6];
  const float* l1w  = (const float*)d_in[7];
  const float* l1b  = (const float*)d_in[8];
  const float* l2w  = (const float*)d_in[9];
  const float* l2b  = (const float*)d_in[10];
  const float* ln1g = (const float*)d_in[11];
  const float* ln1bb= (const float*)d_in[12];
  const float* ln2g = (const float*)d_in[13];
  const float* ln2bb= (const float*)d_in[14];
  const float* flng = (const float*)d_in[15];
  const float* flnb = (const float*)d_in[16];
  const float* ww   = (const float*)d_in[17];
  const float* wbias= (const float*)d_in[18];

  char* base = (char*)d_ws;
  size_t off = 0;
  auto alloc = [&](size_t bytes) -> void* {
    off = (off + 255) & ~(size_t)255;
    void* p = base + off;
    off += bytes;
    return p;
  };
  const int SB = S_ * B_;  // 4096 rows, seq-first
  float*  x    = (float*) alloc((size_t)SB * D_ * 4);
  bf16_t* xb   = (bf16_t*)alloc((size_t)SB * D_ * 2);
  bf16_t* qkvb = (bf16_t*)alloc((size_t)SB * 3 * D_ * 2);
  bf16_t* qbuf = (bf16_t*)alloc((size_t)B_ * H_ * S_ * HD_ * 2);
  bf16_t* kbuf = (bf16_t*)alloc((size_t)B_ * H_ * S_ * HD_ * 2);
  bf16_t* vtb  = (bf16_t*)alloc((size_t)B_ * H_ * S_ * HD_ * 2);
  bf16_t* ctxb = (bf16_t*)alloc((size_t)SB * D_ * 2);
  float*  tmpf = (float*) alloc((size_t)SB * D_ * 4);
  bf16_t* hb   = (bf16_t*)alloc((size_t)SB * FF_ * 2);
  bf16_t* inwb = (bf16_t*)alloc((size_t)L_ * 3 * D_ * D_ * 2);
  bf16_t* outwb= (bf16_t*)alloc((size_t)L_ * D_ * D_ * 2);
  bf16_t* erb  = (bf16_t*)alloc((size_t)L_ * S_ * HD_ * 2);
  bf16_t* l1wb = (bf16_t*)alloc((size_t)L_ * FF_ * D_ * 2);
  bf16_t* l2wb = (bf16_t*)alloc((size_t)L_ * D_ * FF_ * 2);
  bf16_t* wwb  = (bf16_t*)alloc((size_t)V_ * D_ * 2);

  auto cvt = [&](const float* in, bf16_t* out, size_t n) {
    cvt_bf16_kernel<<<dim3((unsigned)((n + 255) / 256)), 256, 0, stream>>>(in, out, (int)n);
  };
  cvt(inw,  inwb,  (size_t)L_ * 3 * D_ * D_);
  cvt(outw, outwb, (size_t)L_ * D_ * D_);
  cvt(er,   erb,   (size_t)L_ * S_ * HD_);
  cvt(l1w,  l1wb,  (size_t)L_ * FF_ * D_);
  cvt(l2w,  l2wb,  (size_t)L_ * D_ * FF_);
  cvt(ww,   wwb,   (size_t)V_ * D_);

  embed_pe_kernel<<<dim3((unsigned)(SB * D_ / 256)), 256, 0, stream>>>(ids, emb, x, xb);

  auto gemm = [&](const bf16_t* A, const bf16_t* W, const float* bias,
                  float* oF, bf16_t* oB, int M, int N, int K, int relu) {
    dim3 g((unsigned)((N + 63) / 64), (unsigned)(M / 128));
    gemm_bf16_kernel<<<g, 256, 0, stream>>>(A, W, bias, oF, oB, M, N, K, relu);
  };

  for (int i = 0; i < L_; ++i) {
    // QKV projection
    gemm(xb, inwb + (size_t)i * 3 * D_ * D_, inb + (size_t)i * 3 * D_,
         nullptr, qkvb, SB, 3 * D_, D_, 0);
    split_qkv_kernel<<<dim3((unsigned)(B_ * H_ * S_ * HD_ / 256)), 256, 0, stream>>>(
        qkvb, qbuf, kbuf, vtb);
    // flash attention + relative-position skew
    flash_attn_kernel<<<dim3(S_ / 16, B_ * H_), 32, 0, stream>>>(
        qbuf, kbuf, vtb, erb + (size_t)i * S_ * HD_, ctxb);
    // out projection + residual LN1
    gemm(ctxb, outwb + (size_t)i * D_ * D_, outb + (size_t)i * D_,
         tmpf, nullptr, SB, D_, D_, 0);
    add_ln_kernel<<<dim3((unsigned)SB), 256, 0, stream>>>(
        x, tmpf, ln1g + (size_t)i * D_, ln1bb + (size_t)i * D_, xb);
    // FFN
    gemm(xb, l1wb + (size_t)i * FF_ * D_, l1b + (size_t)i * FF_,
         nullptr, hb, SB, FF_, D_, 1);
    gemm(hb, l2wb + (size_t)i * D_ * FF_, l2b + (size_t)i * D_,
         tmpf, nullptr, SB, D_, FF_, 0);
    add_ln_kernel<<<dim3((unsigned)SB), 256, 0, stream>>>(
        x, tmpf, ln2g + (size_t)i * D_, ln2bb + (size_t)i * D_, xb);
  }

  // final LN + vocab projection (N=267, tiles clamped/guarded)
  add_ln_kernel<<<dim3((unsigned)SB), 256, 0, stream>>>(x, nullptr, flng, flnb, xb);
  gemm(xb, wwb, wbias, (float*)d_out, nullptr, SB, V_, D_, 0);
}
